// Encoder_72026601554062
// MI455X (gfx1250) — matrix operand order repaired and checked by reading, not compiled, not run on hardware
//
#include <hip/hip_runtime.h>
#include <math.h>

// ============================================================================
// CDNA5 (gfx1250) bf16-WMMA transformer encoder, round 3.
//  - v_wmma_f32_16x16x32_bf16 everywhere (fp32 accumulate).
//  - All hot-loop tile staging uses GLOBAL_LOAD_ASYNC_TO_LDS_B128 (inline asm,
//    ASYNCcnt-tracked, per CDNA5 ISA) with double-buffered
//    LDS: one barrier per K-step, zero ds_store traffic in GEMM/attn staging.
//  - Weights are pre-transposed to [N][K] at bf16-conversion time, and the V
//    projection writes [b,h,dv,t] directly, so no transposes remain in-loop.
// ============================================================================

#define DEV __device__ __forceinline__

typedef __attribute__((ext_vector_type(16))) __bf16 bf16x16;
typedef __attribute__((ext_vector_type(8)))  float  f32x8;

union FragU { uint4 u[2]; bf16x16 v; };

DEV unsigned short f2bf(float f) {
  union { float f; unsigned u; } c; c.f = f;
  unsigned r = c.u + 0x7FFFu + ((c.u >> 16) & 1u);   // round-to-nearest-even
  return (unsigned short)(r >> 16);
}

DEV f32x8 wmma_bf16(bf16x16 a, bf16x16 b, f32x8 c) {
  return __builtin_amdgcn_wmma_f32_16x16x32_bf16(false, a, false, b,
                                                 (short)0, c, false, false);
}

DEV bf16x16 ld_frag(const unsigned short* p0, const unsigned short* p1) {
  FragU r;
  r.u[0] = *(const uint4*)p0;
  r.u[1] = *(const uint4*)p1;
  return r.v;
}

// ---- CDNA5 async tensor path --------------------------------------------
// LDS byte offset of a __shared__ object == low 32 bits of its flat address.
DEV unsigned lds_off(const void* p) { return (unsigned)(uintptr_t)p; }

// GLOBAL_LOAD_ASYNC_TO_LDS_B128: VDST = per-lane LDS address VGPR,
// VADDR = 64-bit global address (GV mode), tracked by ASYNCcnt.
DEV void async_b128(unsigned lds, const unsigned short* g) {
  asm volatile("global_load_async_to_lds_b128 %0, %1, off"
               :: "v"(lds), "v"(g) : "memory");
}
DEV void wait_asynccnt0() { asm volatile("s_wait_asynccnt 0" ::: "memory"); }

// ============================================================================
// bf16 GEMM:  C[M,N] = A[M,K] * Bt[N,K]^T   (+bias)(+resid)(relu)
// Block: 256x128 tile, BK=32, 256 thr = 8 waves (4 M x 2 N);
// wave = 64x64 macro-tile = 16 WMMA per K-step, one barrier per K-step,
// double-buffered async-to-LDS staging.
// ============================================================================
template<bool BIAS, bool RESID, bool RELU, bool OUTF, bool OUTB, bool OUTVT>
__global__ __launch_bounds__(256)
void gemm_bf16_k(const unsigned short* __restrict__ A,
                 const unsigned short* __restrict__ Bt,
                 const float* __restrict__ bias,
                 const float* __restrict__ resid,
                 float* __restrict__ Cf,
                 unsigned short* __restrict__ Cb,
                 int M, int N, int K)
{
  constexpr int BM = 256, BN = 128, BK = 32;
  constexpr int AS = BK + 8;      // 40-half row stride: 80B, 16B-aligned
  __shared__ __align__(16) unsigned short As[2][BM * AS];   // 2 x 20 KB
  __shared__ __align__(16) unsigned short Bs[2][BN * AS];   // 2 x 10 KB

  const int tid  = threadIdx.x;
  const int lane = tid & 31, wave = tid >> 5;
  const int wm = wave & 3, wn = wave >> 2;
  const int hlo = lane & 15, hhi = lane >> 4;
  const int tiles_n = N / BN;
  const int bm = (blockIdx.x / tiles_n) * BM;
  const int bn = (blockIdx.x % tiles_n) * BN;

  // Async-stage one K-tile: A = 256 rows x 4 b128-segs, B = 128 rows x 4 segs.
  auto stage = [&](int buf, int k0) {
#pragma unroll
    for (int i = 0; i < 4; i++) {
      int s = tid + i * 256, row = s >> 2, seg = s & 3;
      async_b128(lds_off(&As[buf][row * AS + seg * 8]),
                 A + (size_t)(bm + row) * K + k0 + seg * 8);
    }
#pragma unroll
    for (int i = 0; i < 2; i++) {
      int s = tid + i * 256, row = s >> 2, seg = s & 3;
      async_b128(lds_off(&Bs[buf][row * AS + seg * 8]),
                 Bt + (size_t)(bn + row) * K + k0 + seg * 8);
    }
  };

  f32x8 acc[4][4];
#pragma unroll
  for (int mt = 0; mt < 4; mt++)
#pragma unroll
    for (int nt = 0; nt < 4; nt++)
#pragma unroll
      for (int j = 0; j < 8; j++) acc[mt][nt][j] = 0.f;

  stage(0, 0);
  wait_asynccnt0();
  __syncthreads();

  int buf = 0;
  for (int k0 = 0; k0 < K; k0 += BK) {
    if (k0 + BK < K) stage(buf ^ 1, k0 + BK);   // overlap DMA with WMMA

    bf16x16 af[4], bfr[4];
#pragma unroll
    for (int mt = 0; mt < 4; mt++) {
      const unsigned short* rp = &As[buf][(wm * 64 + mt * 16 + hlo) * AS];
      af[mt] = ld_frag(rp + hhi * 8, rp + 16 + hhi * 8);
    }
#pragma unroll
    for (int nt = 0; nt < 4; nt++) {
      const unsigned short* rp = &Bs[buf][(wn * 64 + nt * 16 + hlo) * AS];
      bfr[nt] = ld_frag(rp + hhi * 8, rp + 16 + hhi * 8);
    }
#pragma unroll
    for (int mt = 0; mt < 4; mt++)
#pragma unroll
      for (int nt = 0; nt < 4; nt++)
        acc[mt][nt] = wmma_bf16(af[mt], bfr[nt], acc[mt][nt]);

    wait_asynccnt0();
    __syncthreads();
    buf ^= 1;
  }

  // ---- epilogue (C layout: M = j + (lane/16)*8, N = lane%16) ----
#pragma unroll
  for (int mt = 0; mt < 4; mt++) {
#pragma unroll
    for (int nt = 0; nt < 4; nt++) {
      const int n  = bn + wn * 64 + nt * 16 + hlo;
      const int m0 = bm + wm * 64 + mt * 16 + hhi * 8;
      float bv = BIAS ? bias[n] : 0.f;
#pragma unroll
      for (int j = 0; j < 8; j++) {
        const int m = m0 + j;
        const size_t off = (size_t)m * N + n;
        float v = acc[mt][nt][j] + bv;
        if (RESID) v += resid[off];
        if (RELU)  v = fmaxf(v, 0.f);
        if (OUTF)  Cf[off] = v;
        if (OUTB) {
          // OUTVT: write V output as [b, h, dv, t]  (T=1024, H=12, DV=64)
          size_t ob = OUTVT
            ? ((((size_t)(m >> 10) * 12 + (n >> 6)) * 64 + (n & 63)) * 1024
               + (size_t)(m & 1023))
            : off;
          Cb[ob] = f2bf(v);
        }
      }
    }
  }
}

// ============================================================================
// Flash attention. Grid = B*H*(T/128). Block = 128 thr = 4 waves x 32 q-rows.
// Double-buffered async K/V tiles (64 keys/block), one barrier per block.
// K tile row-major == K^T B-operand; V arrives pre-transposed [dv][t].
// ============================================================================
__global__ __launch_bounds__(128)
void attn_k(const unsigned short* __restrict__ Q,
            const unsigned short* __restrict__ Km,
            const unsigned short* __restrict__ VT,
            const unsigned char* __restrict__ mask,
            unsigned short* __restrict__ O,
            float scale)
{
  constexpr int T = 1024, H = 12, DStr = 72, QB = 128;
  __shared__ __align__(16) unsigned short Ks[2][64 * DStr];
  __shared__ __align__(16) unsigned short Vs[2][64 * DStr];
  __shared__ __align__(16) unsigned short Ps[4 * 32 * DStr];
  __shared__ float smadd[2][64];

  const int tid = threadIdx.x, lane = tid & 31, wave = tid >> 5;
  const int hlo = lane & 15, hhi = lane >> 4;
  const int nqb = T / QB;
  const int qblk = blockIdx.x % nqb;
  const int h = (blockIdx.x / nqb) % H;
  const int b = blockIdx.x / (nqb * H);
  const size_t rowstr = (size_t)H * 64;                 // 768
  const unsigned short* Qb  = Q  + ((size_t)b * T) * rowstr + h * 64;
  const unsigned short* Kb  = Km + ((size_t)b * T) * rowstr + h * 64;
  const unsigned short* VTb = VT + ((size_t)(b * H + h)) * 64 * T;   // [dv][t]
  const int q0 = qblk * QB + wave * 32;

  // Async-stage one 64-key block into buffer `buf`.
  auto stage = [&](int buf, int s0) {
#pragma unroll
    for (int i = 0; i < 4; i++) {
      int s = tid + i * 128, r = s >> 3, seg = s & 7;
      async_b128(lds_off(&Ks[buf][r * DStr + seg * 8]),
                 Kb + (size_t)(s0 + r) * rowstr + seg * 8);
      async_b128(lds_off(&Vs[buf][r * DStr + seg * 8]),
                 VTb + (size_t)r * T + s0 + seg * 8);
    }
    if (tid < 64)
      smadd[buf][tid] = mask[(size_t)b * T + s0 + tid] ? 0.f : -1e30f;
  };

  // Q fragments resident in registers: 2 M-tiles x 2 K-steps.
  bf16x16 qf[2][2];
#pragma unroll
  for (int mt = 0; mt < 2; mt++)
#pragma unroll
    for (int ks = 0; ks < 2; ks++) {
      const unsigned short* rp = Qb + (size_t)(q0 + mt * 16 + hlo) * rowstr + ks * 32;
      qf[mt][ks] = ld_frag(rp + hhi * 8, rp + 16 + hhi * 8);
    }

  f32x8 acc[2][4];
  float mst[2][8], lst[2][8];
#pragma unroll
  for (int mt = 0; mt < 2; mt++)
#pragma unroll
    for (int j = 0; j < 8; j++) { mst[mt][j] = -1e30f; lst[mt][j] = 0.f; }
#pragma unroll
  for (int mt = 0; mt < 2; mt++)
#pragma unroll
    for (int nt = 0; nt < 4; nt++)
#pragma unroll
      for (int j = 0; j < 8; j++) acc[mt][nt][j] = 0.f;

  unsigned short* pw = Ps + wave * 32 * DStr;

  stage(0, 0);
  wait_asynccnt0();
  __syncthreads();

  int buf = 0;
  for (int s0 = 0; s0 < T; s0 += 64) {
    if (s0 + 64 < T) stage(buf ^ 1, s0 + 64);   // overlap DMA with compute

    // ---- S = Q K^T ----
    f32x8 S[2][4];
#pragma unroll
    for (int mt = 0; mt < 2; mt++)
#pragma unroll
      for (int nt = 0; nt < 4; nt++)
#pragma unroll
        for (int j = 0; j < 8; j++) S[mt][nt][j] = 0.f;
#pragma unroll
    for (int ks = 0; ks < 2; ks++) {
      bf16x16 kf[4];
#pragma unroll
      for (int nt = 0; nt < 4; nt++) {
        const unsigned short* rp = &Ks[buf][(nt * 16 + hlo) * DStr + ks * 32];
        kf[nt] = ld_frag(rp + hhi * 8, rp + 16 + hhi * 8);
      }
#pragma unroll
      for (int mt = 0; mt < 2; mt++)
#pragma unroll
        for (int nt = 0; nt < 4; nt++)
          S[mt][nt] = wmma_bf16(qf[mt][ks], kf[nt], S[mt][nt]);
    }

    // ---- online softmax; P (bf16) to per-wave LDS slab ----
#pragma unroll
    for (int mt = 0; mt < 2; mt++) {
#pragma unroll
      for (int j = 0; j < 8; j++) {
#pragma unroll
        for (int nt = 0; nt < 4; nt++)
          S[mt][nt][j] = S[mt][nt][j] * scale + smadd[buf][nt * 16 + hlo];
        float vmax = fmaxf(fmaxf(S[mt][0][j], S[mt][1][j]),
                           fmaxf(S[mt][2][j], S[mt][3][j]));
        for (int off = 1; off < 16; off <<= 1)
          vmax = fmaxf(vmax, __shfl_xor(vmax, off, 16));
        float nm   = fmaxf(mst[mt][j], vmax);
        float corr = __expf(mst[mt][j] - nm);
        float rsum = 0.f;
#pragma unroll
        for (int nt = 0; nt < 4; nt++) {
          float p = __expf(S[mt][nt][j] - nm);
          S[mt][nt][j] = p; rsum += p;
        }
        for (int off = 1; off < 16; off <<= 1)
          rsum += __shfl_xor(rsum, off, 16);
        lst[mt][j] = lst[mt][j] * corr + rsum;
        mst[mt][j] = nm;
#pragma unroll
        for (int nt = 0; nt < 4; nt++) acc[mt][nt][j] *= corr;
        const int prow = mt * 16 + hhi * 8 + j;
#pragma unroll
        for (int nt = 0; nt < 4; nt++)
          pw[prow * DStr + nt * 16 + hlo] = f2bf(S[mt][nt][j]);
      }
    }
    // per-wave P slab RAW: compiler inserts s_wait_dscnt; no barrier needed.

    // ---- acc += P * V ----
#pragma unroll
    for (int ks = 0; ks < 2; ks++) {
      bf16x16 pf[2], vf[4];
#pragma unroll
      for (int mt = 0; mt < 2; mt++) {
        const unsigned short* rp = pw + (mt * 16 + hlo) * DStr + ks * 32;
        pf[mt] = ld_frag(rp + hhi * 8, rp + 16 + hhi * 8);
      }
#pragma unroll
      for (int nt = 0; nt < 4; nt++) {
        const unsigned short* rp = &Vs[buf][(nt * 16 + hlo) * DStr + ks * 32];
        vf[nt] = ld_frag(rp + hhi * 8, rp + 16 + hhi * 8);
      }
#pragma unroll
      for (int mt = 0; mt < 2; mt++)
#pragma unroll
        for (int nt = 0; nt < 4; nt++)
          acc[mt][nt] = wmma_bf16(pf[mt], vf[nt], acc[mt][nt]);
    }

    wait_asynccnt0();
    __syncthreads();
    buf ^= 1;
  }

  unsigned short* Ob = O + ((size_t)b * T) * rowstr + h * 64;
#pragma unroll
  for (int mt = 0; mt < 2; mt++)
#pragma unroll
    for (int nt = 0; nt < 4; nt++)
#pragma unroll
      for (int j = 0; j < 8; j++) {
        int t  = q0 + mt * 16 + hhi * 8 + j;
        int dv = nt * 16 + hlo;
        float v = acc[mt][nt][j] / fmaxf(lst[mt][j], 1e-30f);
        Ob[(size_t)t * rowstr + dv] = f2bf(v);
      }
}

// ============================================================================
// LayerNorm: one wave per row (DM=768, 24 elems/lane), emits fp32 + bf16.
// ============================================================================
__global__ __launch_bounds__(256)
void ln_k(const float* __restrict__ X, const float* __restrict__ g,
          const float* __restrict__ bt, float* __restrict__ Yf,
          unsigned short* __restrict__ Yb)
{
  constexpr int DM = 768;
  const int lane = threadIdx.x & 31, wave = threadIdx.x >> 5;
  const size_t row = (size_t)blockIdx.x * 8 + wave;
  const float* x = X + row * DM;
  float v[24]; float s = 0.f, s2 = 0.f;
#pragma unroll
  for (int i = 0; i < 24; i++) {
    v[i] = x[lane + i * 32]; s += v[i]; s2 += v[i] * v[i];
  }
  for (int off = 1; off < 32; off <<= 1) {
    s  += __shfl_xor(s,  off, 32);
    s2 += __shfl_xor(s2, off, 32);
  }
  float mu  = s  * (1.f / DM);
  float var = s2 * (1.f / DM) - mu * mu;
  float r = rsqrtf(var + 1e-5f);
#pragma unroll
  for (int i = 0; i < 24; i++) {
    int d = lane + i * 32;
    float o = (v[i] - mu) * r * g[d] + bt[d];
    Yf[row * DM + d] = o;
    Yb[row * DM + d] = f2bf(o);
  }
}

// ============================================================================
__global__ void posembed_k(const float* __restrict__ Xin, float* __restrict__ Xf,
                           unsigned short* __restrict__ Xb, int n)
{
  constexpr int DM = 768, T = 1024;
  int i = blockIdx.x * blockDim.x + threadIdx.x;
  if (i >= n) return;
  int d = i % DM;
  int t = (i / DM) % T;
  float ex  = (float)((d >> 1) << 1) * (1.f / DM);
  float ang = (float)t * __powf(10000.f, -ex);
  float pe  = ((d & 1) == 0) ? __sinf(ang) : __cosf(ang);
  float v = Xin[i] + pe;
  Xf[i] = v; Xb[i] = f2bf(v);
}

// Generic fp32 [K][N] -> bf16 transposed [N][K]
__global__ void convbt_k(const float* __restrict__ src,
                         unsigned short* __restrict__ dst, int K, int N)
{
  int i = blockIdx.x * blockDim.x + threadIdx.x;
  if (i >= K * N) return;
  int n = i / K, k = i % K;
  dst[i] = f2bf(src[(size_t)k * N + n]);
}

// wq/wk/wv [H, DM, DK] fp32 -> Bt layout [n = h*64+kk][d] bf16
__global__ void convqkv_k(const float* __restrict__ src,
                          unsigned short* __restrict__ dst)
{
  constexpr int DM = 768, DK = 64, n_el = DM * 768;
  int i = blockIdx.x * blockDim.x + threadIdx.x;
  if (i >= n_el) return;
  int n = i / 768, d = i % 768;          // dst[n*768 + d]
  int h = n >> 6, kk = n & 63;
  dst[i] = f2bf(src[(size_t)(h * DM + d) * DK + kk]);
}

// ============================================================================
extern "C" void kernel_launch(void* const* d_in, const int* in_sizes, int n_in,
                              void* d_out, int out_size, void* d_ws, size_t ws_size,
                              hipStream_t stream)
{
  (void)in_sizes; (void)n_in; (void)out_size; (void)ws_size;
  const float* x_in   = (const float*)d_in[0];
  const unsigned char* mask = (const unsigned char*)d_in[1];
  const float* wq     = (const float*)d_in[2];
  const float* wk     = (const float*)d_in[3];
  const float* wv     = (const float*)d_in[4];
  const float* proj_w = (const float*)d_in[5];
  const float* proj_b = (const float*)d_in[6];
  const float* ln1_g  = (const float*)d_in[7];
  const float* ln1_b  = (const float*)d_in[8];
  const float* w1     = (const float*)d_in[9];
  const float* b1     = (const float*)d_in[10];
  const float* w2     = (const float*)d_in[11];
  const float* b2     = (const float*)d_in[12];
  const float* ln2_g  = (const float*)d_in[13];
  const float* ln2_b  = (const float*)d_in[14];

  constexpr int NB = 8, T = 1024, DM = 768, H = 12, DK = 64, DH = 3072, NL = 6;
  const size_t NT = (size_t)NB * T;                 // 8192 tokens

  char* wsp = (char*)d_ws;
  auto carve = [&](size_t bytes) {
    void* p = wsp; wsp += (bytes + 255) & ~(size_t)255; return p;
  };
  float*          xf  = (float*)         carve(NT * DM * 4);
  float*          t0  = (float*)         carve(NT * DM * 4);
  unsigned short* xb  = (unsigned short*)carve(NT * DM * 2);
  unsigned short* hb  = (unsigned short*)carve(NT * DH * 2);
  unsigned short* qb  = (unsigned short*)carve(NT * DM * 2);
  unsigned short* kb  = (unsigned short*)carve(NT * DM * 2);
  unsigned short* vtb = (unsigned short*)carve(NT * DM * 2);   // [b,h,dv,t]
  unsigned short* ob  = (unsigned short*)carve(NT * DM * 2);
  unsigned short* wqb = (unsigned short*)carve((size_t)DM * DM * 2);
  unsigned short* wkb = (unsigned short*)carve((size_t)DM * DM * 2);
  unsigned short* wvb = (unsigned short*)carve((size_t)DM * DM * 2);
  unsigned short* wpb = (unsigned short*)carve((size_t)DM * DM * 2);
  unsigned short* w1b = (unsigned short*)carve((size_t)DM * DH * 2);
  unsigned short* w2b = (unsigned short*)carve((size_t)DH * DM * 2);

  const int n_act = (int)(NT * DM);
  posembed_k<<<(n_act + 255) / 256, 256, 0, stream>>>(x_in, xf, xb, n_act);

  const int NW  = DM * DM;        // 589824
  const int NW1 = DM * DH;        // 2359296
  const float scale = 1.0f / sqrtf((float)DM);
  const dim3 g_sm((8192 / 256) * (768 / 128));    // 192 tiles
  const dim3 g_lg((8192 / 256) * (3072 / 128));   // 768 tiles

  for (int l = 0; l < NL; l++) {
    convqkv_k<<<(NW + 255) / 256, 256, 0, stream>>>(wq + (size_t)l * NW, wqb);
    convqkv_k<<<(NW + 255) / 256, 256, 0, stream>>>(wk + (size_t)l * NW, wkb);
    convqkv_k<<<(NW + 255) / 256, 256, 0, stream>>>(wv + (size_t)l * NW, wvb);
    convbt_k<<<(NW + 255) / 256, 256, 0, stream>>>(proj_w + (size_t)l * NW, wpb, DM, DM);
    convbt_k<<<(NW1 + 255) / 256, 256, 0, stream>>>(w1 + (size_t)l * NW1, w1b, DM, DH);
    convbt_k<<<(NW1 + 255) / 256, 256, 0, stream>>>(w2 + (size_t)l * NW1, w2b, DH, DM);

    // Q/K: normal [m][n] bf16 out; V: [b,h,dv,t] transposed bf16 out.
    gemm_bf16_k<false, false, false, false, true, false><<<g_sm, 256, 0, stream>>>(
        xb, wqb, nullptr, nullptr, nullptr, qb, 8192, 768, 768);
    gemm_bf16_k<false, false, false, false, true, false><<<g_sm, 256, 0, stream>>>(
        xb, wkb, nullptr, nullptr, nullptr, kb, 8192, 768, 768);
    gemm_bf16_k<false, false, false, false, true, true><<<g_sm, 256, 0, stream>>>(
        xb, wvb, nullptr, nullptr, nullptr, vtb, 8192, 768, 768);

    attn_k<<<NB * H * (T / 128), 128, 0, stream>>>(qb, kb, vtb, mask, ob, scale);

    gemm_bf16_k<true, true, false, true, false, false><<<g_sm, 256, 0, stream>>>(
        ob, wpb, proj_b + (size_t)l * DM, xf, t0, nullptr, 8192, 768, 768);
    ln_k<<<NT / 8, 256, 0, stream>>>(t0, ln1_g + (size_t)l * DM,
                                     ln1_b + (size_t)l * DM, xf, xb);

    gemm_bf16_k<true, false, true, false, true, false><<<g_lg, 256, 0, stream>>>(
        xb, w1b, b1 + (size_t)l * DH, nullptr, nullptr, hb, 8192, 3072, 768);
    gemm_bf16_k<true, true, false, true, false, false><<<g_sm, 256, 0, stream>>>(
        hb, w2b, b2 + (size_t)l * DM, xf, t0, nullptr, 8192, 768, 3072);
    ln_k<<<NT / 8, 256, 0, stream>>>(t0, ln2_g + (size_t)l * DM,
                                     ln2_b + (size_t)l * DM, xf, xb);
  }

  hipMemcpyAsync(d_out, xf, NT * DM * sizeof(float),
                 hipMemcpyDeviceToDevice, stream);
}
